// SelfAttention_74981539054039
// MI455X (gfx1250) — matrix operand order repaired and checked by reading, not compiled, not run on hardware
//
#include <hip/hip_runtime.h>
#include <hip/hip_bf16.h>

#define BATCH 4
#define CDIM  256
#define LDIM  32768
#define YLB   128   // l-columns per workgroup in the y kernel

typedef __attribute__((ext_vector_type(16))) __bf16 v16bf;
typedef __attribute__((ext_vector_type(8)))  float  v8f;

union BF16x16 {
    unsigned u32[8];
    v16bf v;
};

// pack two f32 into a bf16 pair with one v_perm_b32 (truncating round)
__device__ __forceinline__ unsigned bfpair(float lo, float hi) {
    return __builtin_amdgcn_perm(__float_as_uint(hi), __float_as_uint(lo), 0x07060302u);
}

// 16 contiguous floats -> v16bf  (B-fragment: K-contiguous per lane)
__device__ __forceinline__ v16bf pack_contig16(const float* p) {
    const float4* q = (const float4*)p;
    float4 a = q[0], b = q[1], c = q[2], d = q[3];
    BF16x16 r;
    r.u32[0] = bfpair(a.x, a.y); r.u32[1] = bfpair(a.z, a.w);
    r.u32[2] = bfpair(b.x, b.y); r.u32[3] = bfpair(b.z, b.w);
    r.u32[4] = bfpair(c.x, c.y); r.u32[5] = bfpair(c.z, c.w);
    r.u32[6] = bfpair(d.x, d.y); r.u32[7] = bfpair(d.z, d.w);
    return r.v;
}

// floats p[0..7] and p[16..23] -> v16bf (A-fragment split-K layout)
__device__ __forceinline__ v16bf pack_split8(const float* p) {
    const float4* q = (const float4*)p;
    float4 a = q[0], b = q[1], c = q[4], d = q[5];
    BF16x16 r;
    r.u32[0] = bfpair(a.x, a.y); r.u32[1] = bfpair(a.z, a.w);
    r.u32[2] = bfpair(b.x, b.y); r.u32[3] = bfpair(b.z, b.w);
    r.u32[4] = bfpair(c.x, c.y); r.u32[5] = bfpair(c.z, c.w);
    r.u32[6] = bfpair(d.x, d.y); r.u32[7] = bfpair(d.z, d.w);
    return r.v;
}

// ---- CDNA5 async copy: global -> LDS, tracked by ASYNCcnt ------------------
__device__ __forceinline__ void async_load_b128(const float* g, void* lds) {
    asm volatile("global_load_async_to_lds_b128 %0, %1, off"
                 :: "v"((unsigned)(uintptr_t)lds),
                    "v"((unsigned long long)(uintptr_t)g)
                 : "memory");
}
__device__ __forceinline__ void async_wait0() {
    asm volatile("s_wait_asynccnt 0x0" ::: "memory");
}
// ---------------------------------------------------------------------------

__global__ void zero_kernel(float* __restrict__ p, int n) {
    int i = blockIdx.x * blockDim.x + threadIdx.x;
    if (i < n) p[i] = 0.f;
}

// G[b] += x[b] @ x[b]^T over an L-split. 4 waves/WG, each wave owns a 16x64
// strip of a 64x64 G block; bf16 WMMA, fp32 accumulate, atomic reduction.
__global__ __launch_bounds__(128) void gram_kernel(const float* __restrict__ x,
                                                   float* __restrict__ G) {
    const int wg  = blockIdx.x;          // BATCH * 8 * 16 workgroups
    const int b   = wg >> 7;
    const int rem = wg & 127;
    const int ls  = rem >> 4;            // L-split 0..7 (4096 each)
    const int blk = rem & 15;            // 4x4 grid of 64x64 blocks
    const int m0  = (blk >> 2) << 6;
    const int n0  = (blk & 3)  << 6;
    const int wave = threadIdx.x >> 5;
    const int lane = threadIdx.x & 31;
    const int l15  = lane & 15;
    const int hi   = lane >> 4;

    const float* xb = x + (size_t)b * CDIM * LDIM;
    const size_t arowOff = (size_t)(m0 + wave * 16 + l15) * LDIM;
    const int kbA = hi ? 16 : 0;         // A: lanes16-31 hold K=16..31
    const int kbB = hi ? 16 : 0;         // B: lanes16-31 hold K=16..31

    size_t bOff[4];
#pragma unroll
    for (int t = 0; t < 4; ++t)
        bOff[t] = (size_t)(n0 + t * 16 + l15) * LDIM;

    v8f acc[4] = {};
    const int lbeg = ls << 12, lend = lbeg + 4096;
    for (int l0 = lbeg; l0 < lend; l0 += 32) {
        v16bf a = pack_contig16(xb + arowOff + l0 + kbA);
#pragma unroll
        for (int t = 0; t < 4; ++t) {
            v16bf bt = pack_contig16(xb + bOff[t] + l0 + kbB);
            acc[t] = __builtin_amdgcn_wmma_f32_16x16x32_bf16(
                false, a, false, bt, (short)0, acc[t], false, false);
        }
    }

    float* Gb = G + ((size_t)b << 16);
    const int rbase = m0 + wave * 16 + (hi << 3);
#pragma unroll
    for (int r = 0; r < 8; ++r) {
        const int row = (rbase + r) * CDIM + n0 + l15;
#pragma unroll
        for (int t = 0; t < 4; ++t)
            atomicAdd(&Gb[row + t * 16], acc[t][r]);
    }
}

// Small fp32 matmul: C[b][row,j] = sum_k A[b][row,k] * (TRANSB ? B[b][j,k] : B[b][k,j])
template <int TRANSB>
__global__ __launch_bounds__(256) void mm256_kernel(const float* __restrict__ A, long aStride,
                                                    const float* __restrict__ Bm, long bStride,
                                                    float* __restrict__ Cm) {
    const int row = blockIdx.x;
    const int b   = blockIdx.y;
    const int tid = threadIdx.x;
    __shared__ float arow[CDIM];
    const float* Ab = A  + (size_t)b * aStride;
    const float* Bb = Bm + (size_t)b * bStride;
    arow[tid] = Ab[row * CDIM + tid];
    __syncthreads();
    float sum = 0.f;
#pragma unroll 8
    for (int k = 0; k < CDIM; ++k)
        sum += arow[k] * (TRANSB ? Bb[tid * CDIM + k] : Bb[k * CDIM + tid]);
    Cm[((size_t)b << 16) + row * CDIM + tid] = sum;
}

// softmax over axis=-2 (the q rows) of A[b, q, k]; block = one (b, k) column.
__global__ __launch_bounds__(256) void softmax_col_kernel(float* __restrict__ A) {
    const int k = blockIdx.x, b = blockIdx.y, q = threadIdx.x;
    __shared__ float red[CDIM];
    float* Ab = A + ((size_t)b << 16);
    float v = Ab[q * CDIM + k];
    red[q] = v; __syncthreads();
    for (int s = 128; s > 0; s >>= 1) {
        if (q < s) red[q] = fmaxf(red[q], red[q + s]);
        __syncthreads();
    }
    float m = red[0]; __syncthreads();
    float e = __expf(v - m);
    red[q] = e; __syncthreads();
    for (int s = 128; s > 0; s >>= 1) {
        if (q < s) red[q] += red[q + s];
        __syncthreads();
    }
    Ab[q * CDIM + k] = e / red[0];
}

// y[b] = M[b] @ x[b]  ([256x256] x [256xL]).
// Double-buffered: async global->LDS staging of 32x128 x tiles overlapped
// with bf16 WMMA compute on the other buffer.
__global__ __launch_bounds__(256) void attn_y_kernel(const float* __restrict__ x,
                                                     const float* __restrict__ Mm,
                                                     float* __restrict__ y) {
    const int lblk = blockIdx.x;   // 0..255
    const int qblk = blockIdx.y;   // 0..1
    const int b    = blockIdx.z;
    const int tid  = threadIdx.x;
    const int wave = tid >> 5, lane = tid & 31, l15 = lane & 15, hi = lane >> 4;
    const int l0 = lblk * YLB;
    const int q0 = qblk * 128;

    __shared__ float tileX[2][32][YLB + 4];   // 2 x 16.9 KB

    const float* xb = x + (size_t)b * CDIM * LDIM;
    const float* Mb = Mm + ((size_t)b << 16);
    const int arow = q0 + wave * 16 + l15;
    const int kbA = hi ? 16 : 0;
    const int kbB = hi ? 16 : 0;

    // per-thread staging slots (4 x b128 per thread = 16 KB per workgroup)
    int sc[4], sl[4];
#pragma unroll
    for (int i = 0; i < 4; ++i) {
        int idx4 = tid + (i << 8);       // 0..1023 float4 slots
        sc[i] = idx4 >> 5;               // row 0..31
        sl[i] = (idx4 & 31) << 2;        // col 0,4,...,124
    }

    v8f acc[8] = {};

    // prologue: stage tile kc=0 into buffer 0
#pragma unroll
    for (int i = 0; i < 4; ++i)
        async_load_b128(xb + (size_t)sc[i] * LDIM + l0 + sl[i], &tileX[0][sc[i]][sl[i]]);
    async_wait0();
    __syncthreads();

    for (int kc = 0; kc < CDIM; kc += 32) {
        const int cur = (kc >> 5) & 1;
        if (kc + 32 < CDIM) {            // stage next tile into other buffer
#pragma unroll
            for (int i = 0; i < 4; ++i)
                async_load_b128(xb + (size_t)(kc + 32 + sc[i]) * LDIM + l0 + sl[i],
                                &tileX[cur ^ 1][sc[i]][sl[i]]);
        }

        v16bf a = pack_contig16(Mb + arow * CDIM + kc + kbA);
#pragma unroll
        for (int t = 0; t < 8; ++t) {
            const int n = t * 16 + l15;
            BF16x16 bb;
#pragma unroll
            for (int j = 0; j < 8; ++j)
                bb.u32[j] = bfpair(tileX[cur][kbB + 2 * j][n],
                                   tileX[cur][kbB + 2 * j + 1][n]);
            acc[t] = __builtin_amdgcn_wmma_f32_16x16x32_bf16(
                false, a, false, bb.v, (short)0, acc[t], false, false);
        }

        async_wait0();
        __syncthreads();
    }

    float* yb = y + (size_t)b * CDIM * LDIM;
#pragma unroll
    for (int t = 0; t < 8; ++t) {
        const int l = l0 + t * 16 + l15;
#pragma unroll
        for (int r = 0; r < 8; ++r) {
            const int q = q0 + wave * 16 + (hi << 3) + r;
            yb[(size_t)q * LDIM + l] = acc[t][r];
        }
    }
}

extern "C" void kernel_launch(void* const* d_in, const int* in_sizes, int n_in,
                              void* d_out, int out_size, void* d_ws, size_t ws_size,
                              hipStream_t stream) {
    const float* x  = (const float*)d_in[0];
    const float* WQ = (const float*)d_in[1];
    const float* WK = (const float*)d_in[2];
    const float* WV = (const float*)d_in[3];
    float* y = (float*)d_out;

    // workspace: 4 per-batch 256x256 fp32 matrices (4 MB total)
    float* G  = (float*)d_ws;             // Gram  x x^T
    float* T1 = G  + BATCH * 65536;       // W_Q @ G
    float* Am = T1 + BATCH * 65536;       // A_ then softmax(A_) in place
    float* Mm = Am + BATCH * 65536;       // A @ W_V

    zero_kernel<<<dim3((BATCH * 65536 + 255) / 256), 256, 0, stream>>>(G, BATCH * 65536);
    gram_kernel<<<dim3(BATCH * 8 * 16), 128, 0, stream>>>(x, G);
    mm256_kernel<0><<<dim3(CDIM, BATCH), 256, 0, stream>>>(WQ, 0, G, 65536, T1);
    mm256_kernel<1><<<dim3(CDIM, BATCH), 256, 0, stream>>>(T1, 65536, WK, 0, Am);
    softmax_col_kernel<<<dim3(CDIM, BATCH), 256, 0, stream>>>(Am);
    mm256_kernel<0><<<dim3(CDIM, BATCH), 256, 0, stream>>>(Am, 65536, WV, 0, Mm);
    attn_y_kernel<<<dim3(LDIM / YLB, 2, BATCH), 256, 0, stream>>>(x, Mm, y);
}
